// Masker_FFT_34909494182128
// MI455X (gfx1250) — matrix operand order repaired and checked by reading, not compiled, not run on hardware
//
#include <hip/hip_runtime.h>
#include <hip/hip_bf16.h>

typedef __attribute__((ext_vector_type(16))) __bf16 v16bf;
typedef __attribute__((ext_vector_type(8)))  float  v8f;
typedef __attribute__((ext_vector_type(4)))  int    v4i;
typedef __attribute__((ext_vector_type(2)))  __bf16 v2bf;

#define SPLIT   392
#define BN_EPS  1e-5f

#define AS1 __attribute__((address_space(1)))
#define AS3 __attribute__((address_space(3)))

#if __has_builtin(__builtin_amdgcn_global_load_async_to_lds_b128)
#define USE_ASYNC_LDS 1
#define ASYNC_CP_B128(gp, lp) \
    __builtin_amdgcn_global_load_async_to_lds_b128((AS1 v4i*)(void*)(gp), (AS3 v4i*)(void*)(lp), 0, 0)
#else
#define USE_ASYNC_LDS 0
#endif

#if __has_builtin(__builtin_amdgcn_s_wait_asynccnt)
#define WAIT_ASYNC(n) __builtin_amdgcn_s_wait_asynccnt(n)
#else
#define WAIT_ASYNC(n) asm volatile("s_wait_asynccnt " #n ::: "memory")
#endif

// Single f32 -> bf16 (round-half-up; 2 VALU ops)
__device__ __forceinline__ unsigned short f2bf(float f) {
    return (unsigned short)((__float_as_uint(f) + 0x8000u) >> 16);
}

#if __has_builtin(__builtin_amdgcn_cvt_pk_bf16_f32)
__device__ __forceinline__ unsigned int pack2bf(float lo, float hi) {
    v2bf p = __builtin_amdgcn_cvt_pk_bf16_f32(lo, hi);   // v_cvt_pk_bf16_f32 (RNE)
    unsigned int r;
    __builtin_memcpy(&r, &p, 4);
    return r;
}
#else
// 3 VALU ops: 2x v_add_nc_u32 + 1x v_perm_b32 (bytes {7,6} of hi, {3,2} of lo)
__device__ __forceinline__ unsigned int pack2bf(float lo, float hi) {
    unsigned int l = __float_as_uint(lo) + 0x8000u;      // round-half-up
    unsigned int h = __float_as_uint(hi) + 0x8000u;
    return __builtin_amdgcn_perm(h, l, 0x07060302u);
}
#endif

__device__ __forceinline__ unsigned int lowbias32(unsigned int x) {
    x ^= x >> 16; x *= 0x7feb352dU;
    x ^= x >> 15; x *= 0x846ca68bU;
    x ^= x >> 16; return x;
}

// ---------------------------------------------------------------------------
// Stage 1: Re(FFT2(f)) per (b,c) 8x8 tile -> x[64,2048] bf16 (duplicated)
// ---------------------------------------------------------------------------
__global__ void fft_real_to_x(const float* __restrict__ f, unsigned short* __restrict__ X) {
    __shared__ float tile[64];
    const int bc = blockIdx.x;        // b*16 + c
    const int t  = threadIdx.x;       // 0..63, output index u*8+v
    tile[t] = f[bc * 64 + t];
    __syncthreads();
    const float ct[8] = {1.f, 0.70710678f, 0.f, -0.70710678f,
                         -1.f, -0.70710678f, 0.f, 0.70710678f}; // cos(2*pi*k/8)
    const int u = t >> 3, v = t & 7;
    float acc = 0.f;
#pragma unroll
    for (int h = 0; h < 8; ++h)
#pragma unroll
        for (int w = 0; w < 8; ++w)
            acc += tile[h * 8 + w] * ct[(u * h + v * w) & 7];
    const int b = bc >> 4, c = bc & 15;
    const unsigned short hv = f2bf(acc);
    X[b * 2048 + c * 64 + t]        = hv;
    X[b * 2048 + 1024 + c * 64 + t] = hv;
}

// ---------------------------------------------------------------------------
// Stage 2: Y[64,N] = act(bn(X @ W^T + b)).
// Block = 64x64 output strip, BK = 64. 8 waves: 4 M-tiles x 2 N-groups,
// each wave holds 2 accumulators (shared A fragment) -> 4 WMMA / k-loop body.
// W is register-double-buffered (fp32->bf16 pack via v_perm_b32); X tile
// goes global->LDS via global_load_async_to_lds_b128, double-buffered LDS.
// ---------------------------------------------------------------------------
__global__ void __launch_bounds__(256)
gemm_bn_act(const unsigned short* __restrict__ X,   // bf16 [64,K]
            const float* __restrict__ W,            // f32  [N,K]
            const float* __restrict__ bias,
            const float* __restrict__ gamma,        // null => affine=False
            const float* __restrict__ beta,
            const float* __restrict__ rm,
            const float* __restrict__ rv,
            void* __restrict__ Y,
            int K, int N, int relu, int out_bf16)
{
#if USE_ASYNC_LDS
    __shared__ __align__(16) unsigned int sX[2][2048]; // 2 x (64 rows x 32 dwords)
#else
    __shared__ __align__(16) unsigned int sX[1][2048];
#endif
    __shared__ __align__(16) unsigned int sW[2048];    // 64 n-rows x 32 dwords (bf16)

    const int tid   = threadIdx.x;
    const int lane  = tid & 31;
    const int wave  = tid >> 5;
    const int mbase = (wave & 3) << 4;   // 0,16,32,48
    const int ngrp  = (wave >> 2) << 5;  // 0 or 32 (two 16-wide tiles each)
    const int nblk  = blockIdx.x << 6;   // 64-wide N strip

    const unsigned int* X32 = (const unsigned int*)X;
    const int Kw = K >> 1;

    // staging split: thread -> row sr (0..63), quarter sq (0..3): 16 elems of K
    const int sr = tid >> 2;
    const int sq = tid & 3;

    v8f acc0 = {}, acc1 = {};

    // ---- prologue: prefetch tile k0=0
    const float* wp = W + (size_t)(nblk + sr) * (size_t)K + (size_t)(sq * 16);
    float4 wv0 = ((const float4*)wp)[0];
    float4 wv1 = ((const float4*)wp)[1];
    float4 wv2 = ((const float4*)wp)[2];
    float4 wv3 = ((const float4*)wp)[3];

#if USE_ASYNC_LDS
    {
        const unsigned short* g = X + (size_t)sr * K + sq * 16;
        unsigned int* l = &sX[0][sr * 32 + sq * 8];
        ASYNC_CP_B128(g,     l);
        ASYNC_CP_B128(g + 8, l + 4);
    }
#else
    uint4 xv0, xv1;
    {
        const uint4* xp = (const uint4*)(X32 + (size_t)sr * Kw + sq * 8);
        xv0 = xp[0]; xv1 = xp[1];
    }
#endif

    int ibuf = 0;
    for (int k0 = 0; k0 < K; k0 += 64) {
        __syncthreads();                 // prior readers of sW / sX buffers done

        // ---- commit staged W tile (pack fp32 -> bf16 in registers, b128 stores)
        {
            uint4 p0, p1;
            p0.x = pack2bf(wv0.x, wv0.y); p0.y = pack2bf(wv0.z, wv0.w);
            p0.z = pack2bf(wv1.x, wv1.y); p0.w = pack2bf(wv1.z, wv1.w);
            p1.x = pack2bf(wv2.x, wv2.y); p1.y = pack2bf(wv2.z, wv2.w);
            p1.z = pack2bf(wv3.x, wv3.y); p1.w = pack2bf(wv3.z, wv3.w);
            uint4* dw = (uint4*)(sW + sr * 32 + sq * 8);
            dw[0] = p0; dw[1] = p1;
        }
#if !USE_ASYNC_LDS
        {
            uint4* dx = (uint4*)(&sX[0][sr * 32 + sq * 8]);
            dx[0] = xv0; dx[1] = xv1;
        }
#endif
        // ---- prefetch next tile (uniform branch; loads issue early, no wait)
        const bool more = (k0 + 64) < K;
        if (more) {
            const float* wn = W + (size_t)(nblk + sr) * (size_t)K + (size_t)(k0 + 64 + sq * 16);
            wv0 = ((const float4*)wn)[0];
            wv1 = ((const float4*)wn)[1];
            wv2 = ((const float4*)wn)[2];
            wv3 = ((const float4*)wn)[3];
#if USE_ASYNC_LDS
            const unsigned short* g = X + (size_t)sr * K + (k0 + 64) + sq * 16;
            unsigned int* l = &sX[ibuf ^ 1][sr * 32 + sq * 8];
            ASYNC_CP_B128(g,     l);
            ASYNC_CP_B128(g + 8, l + 4);
#else
            const uint4* xp = (const uint4*)(X32 + (size_t)sr * Kw + ((k0 + 64) >> 1) + sq * 8);
            xv0 = xp[0]; xv1 = xp[1];
#endif
        }
#if USE_ASYNC_LDS
        if (more) { WAIT_ASYNC(2); }     // retire current tile; keep prefetch in flight
        else      { WAIT_ASYNC(0); }
#endif
        __syncthreads();                 // tile visible to all waves

        // ---- compute: 2 k-substeps x 2 N-tiles = 4 WMMA
        const uint4* sX4 = (const uint4*)(&sX[ibuf][0]);
        const uint4* sW4 = (const uint4*)sW;
        const int hi = lane >> 4;
        const int arow = mbase + (lane & 15);
        const int bcol = ngrp + (lane & 15);
#pragma unroll
        for (int kk = 0; kk < 64; kk += 32) {
            union { v16bf v; uint4 q[2]; } A, B0, B1;
            const int abase = arow * 32 + (kk >> 1);            // dwords
            A.q[0] = sX4[(abase >> 2) + (hi << 1)];
            A.q[1] = sX4[(abase >> 2) + (hi << 1) + 1];
            const int b0 = bcol * 32 + (kk >> 1) + (hi << 3);   // dwords
            B0.q[0] = sW4[(b0 >> 2)];
            B0.q[1] = sW4[(b0 >> 2) + 1];
            const int b1 = b0 + 16 * 32;                        // +16 columns
            B1.q[0] = sW4[(b1 >> 2)];
            B1.q[1] = sW4[(b1 >> 2) + 1];
            acc0 = __builtin_amdgcn_wmma_f32_16x16x32_bf16(false, A.v, false, B0.v,
                                                           (short)0, acc0, false, false);
            acc1 = __builtin_amdgcn_wmma_f32_16x16x32_bf16(false, A.v, false, B1.v,
                                                           (short)0, acc1, false, false);
        }
#if USE_ASYNC_LDS
        ibuf ^= 1;
#endif
    }

    // ---- epilogue: bias + batchnorm (+optional affine) + optional relu
    const int n0 = nblk + ngrp + (lane & 15);
    const int n1 = n0 + 16;
    const float bi0 = bias[n0],            bi1 = bias[n1];
    const float mu0 = rm[n0],              mu1 = rm[n1];
    const float is0 = rsqrtf(rv[n0] + BN_EPS), is1 = rsqrtf(rv[n1] + BN_EPS);
    float g0 = 1.f, be0 = 0.f, g1v = 1.f, be1 = 0.f;
    if (gamma) { g0 = gamma[n0]; be0 = beta[n0]; g1v = gamma[n1]; be1 = beta[n1]; }
    const int mofs = (lane >> 4) << 3;     // +8 for lanes 16..31
#pragma unroll
    for (int r = 0; r < 8; ++r) {
        const int m = mbase + mofs + r;
        float y0 = (acc0[r] + bi0 - mu0) * is0 * g0 + be0;
        float y1 = (acc1[r] + bi1 - mu1) * is1 * g1v + be1;
        if (relu) { y0 = fmaxf(y0, 0.f); y1 = fmaxf(y1, 0.f); }
        if (out_bf16) {
            ((unsigned short*)Y)[(size_t)m * N + n0] = f2bf(y0);
            ((unsigned short*)Y)[(size_t)m * N + n1] = f2bf(y1);
        } else {
            ((float*)Y)[(size_t)m * N + n0] = y0;
            ((float*)Y)[(size_t)m * N + n1] = y1;
        }
    }
}

// ---------------------------------------------------------------------------
// Stage 3: 1024-iteration gumbel-softmax with running max.
// One block per row; row (2048 f32) lives in registers (8/thread x 256).
// ---------------------------------------------------------------------------
__global__ void __launch_bounds__(256)
gumbel_iter(const float* __restrict__ mask, float* __restrict__ out)
{
    __shared__ float red[8];
    const int b    = blockIdx.x;
    const int tid  = threadIdx.x;
    const int lane = tid & 31;
    const int wave = tid >> 5;

    float m[8], z[8];
#pragma unroll
    for (int e = 0; e < 8; ++e) {
        m[e] = mask[b * 2048 + tid + (e << 8)];
        z[e] = 0.f;
    }

    for (int it = 0; it < 1024; ++it) {
        float t[8];
        float lmax = -3.4e38f;
#pragma unroll
        for (int e = 0; e < 8; ++e) {
            unsigned int h = lowbias32(((unsigned int)it * 131072u +
                                        (unsigned int)(b * 2048 + tid + (e << 8))) ^ 0x9E3779B9u);
            float u   = ((float)(h >> 8) + 0.5f) * (1.0f / 16777216.0f); // (0,1)
            float gum = -__logf(-__logf(u));
            t[e] = (m[e] + gum) * 2.0f;  // 1/tau, tau = 0.5
            lmax = fmaxf(lmax, t[e]);
        }
#pragma unroll
        for (int o = 16; o > 0; o >>= 1) lmax = fmaxf(lmax, __shfl_xor(lmax, o, 32));
        if (lane == 0) red[wave] = lmax;
        __syncthreads();
        float rmax = red[0];
#pragma unroll
        for (int i2 = 1; i2 < 8; ++i2) rmax = fmaxf(rmax, red[i2]);
        __syncthreads();

        float lsum = 0.f;
#pragma unroll
        for (int e = 0; e < 8; ++e) { t[e] = __expf(t[e] - rmax); lsum += t[e]; }
#pragma unroll
        for (int o = 16; o > 0; o >>= 1) lsum += __shfl_xor(lsum, o, 32);
        if (lane == 0) red[wave] = lsum;
        __syncthreads();
        float rsum = red[0];
#pragma unroll
        for (int i2 = 1; i2 < 8; ++i2) rsum += red[i2];
        __syncthreads();

        float inv = 1.0f / rsum;
#pragma unroll
        for (int e = 0; e < 8; ++e) { m[e] = t[e] * inv; z[e] = fmaxf(z[e], m[e]); }
    }

    // tuple output: z[:, :392] flat, then z[:, 392:] flat
#pragma unroll
    for (int e = 0; e < 8; ++e) {
        int j = tid + (e << 8);
        if (j < SPLIT) out[b * SPLIT + j] = z[e];
        else           out[64 * SPLIT + b * (2048 - SPLIT) + (j - SPLIT)] = z[e];
    }
}

// ---------------------------------------------------------------------------
extern "C" void kernel_launch(void* const* d_in, const int* in_sizes, int n_in,
                              void* d_out, int out_size, void* d_ws, size_t ws_size,
                              hipStream_t stream) {
    (void)in_sizes; (void)n_in; (void)out_size; (void)ws_size;
    const float* f   = (const float*)d_in[0];
    const float* W1  = (const float*)d_in[1];
    const float* b1  = (const float*)d_in[2];
    const float* g1  = (const float*)d_in[3];
    const float* be1 = (const float*)d_in[4];
    const float* rm1 = (const float*)d_in[5];
    const float* rv1 = (const float*)d_in[6];
    const float* W2  = (const float*)d_in[7];
    const float* b2  = (const float*)d_in[8];
    const float* g2  = (const float*)d_in[9];
    const float* be2 = (const float*)d_in[10];
    const float* rm2 = (const float*)d_in[11];
    const float* rv2 = (const float*)d_in[12];
    const float* W3  = (const float*)d_in[13];
    const float* b3  = (const float*)d_in[14];
    const float* rm3 = (const float*)d_in[15];
    const float* rv3 = (const float*)d_in[16];

    char* ws = (char*)d_ws;
    unsigned short* X    = (unsigned short*)(ws);                   // 64*2048 bf16 = 256KB
    unsigned short* H1   = (unsigned short*)(ws + (256 << 10));     // 64*8192 bf16 = 1MB
    unsigned short* H2   = (unsigned short*)(ws + (256 << 10) + (1 << 20));
    float*          MASK = (float*)(ws + (256 << 10) + (2 << 20));  // 64*2048 f32 = 512KB

    fft_real_to_x<<<1024, 64, 0, stream>>>(f, X);
    gemm_bn_act<<<8192 / 64, 256, 0, stream>>>(X,  W1, b1, g1, be1, rm1, rv1,
                                               (void*)H1, 2048, 8192, 1, 1);
    gemm_bn_act<<<8192 / 64, 256, 0, stream>>>(H1, W2, b2, g2, be2, rm2, rv2,
                                               (void*)H2, 8192, 8192, 1, 1);
    gemm_bn_act<<<2048 / 64, 256, 0, stream>>>(H2, W3, b3, nullptr, nullptr, rm3, rv3,
                                               (void*)MASK, 8192, 2048, 0, 0);
    gumbel_iter<<<64, 256, 0, stream>>>(MASK, (float*)d_out);
}
